// SpectralAttentionLayer_63187558858868
// MI455X (gfx1250) — matrix operand order repaired and checked by reading, not compiled, not run on hardware
//
#include <hip/hip_runtime.h>
#include <hip/hip_bf16.h>

typedef _Float16 v16h __attribute__((ext_vector_type(16)));
typedef _Float16 v8h  __attribute__((ext_vector_type(8)));
typedef float    v8f  __attribute__((ext_vector_type(8)));

#define DM 32
#define NEG_SLOPE 0.2f

static __device__ __forceinline__ float leakyf(float x) { return x > 0.f ? x : NEG_SLOPE * x; }

// relaxed agent-scope fp32 atomic add -> native GLOBAL_ATOMIC_ADD_F32
static __device__ __forceinline__ void atomAddF32(float* p, float v) {
    __hip_atomic_fetch_add(p, v, __ATOMIC_RELAXED, __HIP_MEMORY_SCOPE_AGENT);
}

// monotonic float -> unsigned key (unsigned compare order == float order)
static __device__ __forceinline__ unsigned fkey(float f) {
    unsigned u = __float_as_uint(f);
    return (u & 0x80000000u) ? ~u : (u | 0x80000000u);
}
static __device__ __forceinline__ float fkey_inv(unsigned k) {
    unsigned u = (k & 0x80000000u) ? (k ^ 0x80000000u) : ~k;
    return __uint_as_float(u);
}

// ---------------- degree / norm ----------------
__global__ void k_edge_deg(const int* __restrict__ dst, int* __restrict__ deg, int E) {
    int e = blockIdx.x * blockDim.x + threadIdx.x;
    if (e >= E) return;
    __hip_atomic_fetch_add(&deg[dst[e]], 1, __ATOMIC_RELAXED, __HIP_MEMORY_SCOPE_AGENT);
}

__global__ void k_norm(const int* __restrict__ deg, float* __restrict__ norm, int N) {
    int i = blockIdx.x * blockDim.x + threadIdx.x;
    if (i >= N) return;
    float d = (float)deg[i];
    norm[i] = rsqrtf(d < 1.f ? 1.f : d);
}

// ---------------- elementwise helpers ----------------
__global__ void k_scale_rows(const float* __restrict__ x, const float* __restrict__ norm,
                             float* __restrict__ hn, int N) {
    long t = (long)blockIdx.x * blockDim.x + threadIdx.x;
    if (t >= (long)N * DM) return;
    int i = (int)(t >> 5);
    hn[t] = x[t] * norm[i];
}

__global__ void k_cast_half(const float* __restrict__ src, _Float16* __restrict__ dstp, int n) {
    int t = blockIdx.x * blockDim.x + threadIdx.x;
    if (t >= n) return;
    dstp[t] = (_Float16)src[t];
}

// ---------------- edge scatter: p[dst] += hn[src] ----------------
__global__ void k_edge_scatter(const int* __restrict__ src, const int* __restrict__ dst,
                               const float* __restrict__ hn, float* __restrict__ p, int E) {
    long t = (long)blockIdx.x * blockDim.x + threadIdx.x;
    if (t >= (long)E * DM) return;
    int e = (int)(t >> 5);
    int k = (int)(t & 31);
    int s = src[e], d = dst[e];
    atomAddF32(p + (long)d * DM + k, hn[(long)s * DM + k]);
}

// ---------------- Cheb recurrence (fused) ----------------
__global__ void k_cheb_x1(const float* __restrict__ x0, const float* __restrict__ p,
                          const float* __restrict__ norm, const float* __restrict__ lam,
                          float* __restrict__ x1, float* __restrict__ hn,
                          _Float16* __restrict__ xth, int N) {
    long t = (long)blockIdx.x * blockDim.x + threadIdx.x;
    if (t >= (long)N * DM) return;
    int i = (int)(t >> 5), k = (int)(t & 31);
    float re = 2.f / lam[0];
    float pn = p[t] * norm[i];            // prop(X0)
    float a0 = x0[t];
    float a1 = -re * pn + a0 * (re - 1.f);
    x1[t] = a1;
    hn[t] = a1 * norm[i];                 // input for next prop
    xth[(long)i * 96 + k]      = (_Float16)a0;
    xth[(long)i * 96 + 32 + k] = (_Float16)a1;
}

__global__ void k_cheb_x2(const float* __restrict__ x0, const float* __restrict__ x1,
                          const float* __restrict__ p, const float* __restrict__ norm,
                          const float* __restrict__ lam, _Float16* __restrict__ xth, int N) {
    long t = (long)blockIdx.x * blockDim.x + threadIdx.x;
    if (t >= (long)N * DM) return;
    int i = (int)(t >> 5), k = (int)(t & 31);
    float re = 2.f / lam[0];
    float pn = p[t] * norm[i];            // prop(X1)
    float a2 = -2.f * re * pn + 2.f * (re - 1.f) * x1[t] - x0[t];
    xth[(long)i * 96 + 64 + k] = (_Float16)a2;
}

// ---------------- WMMA GEMM: out(N,32) = act(X(N,K) @ W(32,K)^T + bias) ----------------
__global__ void k_gemm_wmma(const _Float16* __restrict__ X, const _Float16* __restrict__ W,
                            const float* __restrict__ bias,
                            float* __restrict__ outF, _Float16* __restrict__ outH,
                            int N, int K, int act) {
    const int lane = threadIdx.x & 31;
    const int wave = threadIdx.x >> 5;
    const int wavesPerBlk = blockDim.x >> 5;
    const int tile = blockIdx.x * wavesPerBlk + wave;
    const int nTiles = N >> 4;            // N assumed multiple of 16 (100000 = 6250*16)
    if (tile >= nTiles) return;           // wave-uniform: EXEC all-ones below
    const int row0 = tile << 4;
    const int m = lane & 15;
    const int hi = lane >> 4;             // 0: lanes 0-15, 1: lanes 16-31
    const long rowBase = (long)(row0 + m) * K;

    v8f c0 = {};
    v8f c1 = {};
    for (int kc = 0; kc < K; kc += 32) {
        // A fragment (16x32 f16): elems i -> K = kc + hi*16 + i
        const v8h alo = *(const v8h*)(X + rowBase + kc + hi * 16);
        const v8h ahi = *(const v8h*)(X + rowBase + kc + hi * 16 + 8);
        v16h a;
#pragma unroll
        for (int i = 0; i < 8; ++i) { a[i] = alo[i]; a[8 + i] = ahi[i]; }
        // B fragments (32x16 f16): lane n=m, elems i -> K = kc + hi*16 + i
        const v16h b0 = *(const v16h*)(W + (long)(m) * K + kc + hi * 16);
        const v16h b1 = *(const v16h*)(W + (long)(16 + m) * K + kc + hi * 16);
        c0 = __builtin_amdgcn_wmma_f32_16x16x32_f16(false, a, false, b0, (short)0, c0, false, false);
        c1 = __builtin_amdgcn_wmma_f32_16x16x32_f16(false, a, false, b1, (short)0, c1, false, false);
    }
#pragma unroll
    for (int t = 0; t < 2; ++t) {
        v8f c = t ? c1 : c0;
#pragma unroll
        for (int r = 0; r < 8; ++r) {
            int mm = r + hi * 8;
            int nn = t * 16 + m;
            float v = c[r] + bias[nn];
            if (act) v = v > 0.f ? v : 0.f;
            long o = (long)(row0 + mm) * DM + nn;
            if (outF) outF[o] = v;
            if (outH) outH[o] = (_Float16)v;
        }
    }
}

// ---------------- GATv2 edge passes ----------------
__global__ void k_gat_logits(const int* __restrict__ src, const int* __restrict__ dst,
                             const float* __restrict__ fs, const float* __restrict__ fd,
                             const float* __restrict__ attn,
                             float* __restrict__ logits, unsigned* __restrict__ mkey, int E) {
    int e = blockIdx.x * blockDim.x + threadIdx.x;
    if (e >= E) return;
    int s = src[e], d = dst[e];
    const float4* ps = (const float4*)(fs + (long)s * DM);
    const float4* pd = (const float4*)(fd + (long)d * DM);
    const float4* pa = (const float4*)attn;
    float acc = 0.f;
#pragma unroll
    for (int j = 0; j < 8; ++j) {
        float4 a = ps[j], b = pd[j], w = pa[j];
        acc += leakyf(a.x + b.x) * w.x + leakyf(a.y + b.y) * w.y +
               leakyf(a.z + b.z) * w.z + leakyf(a.w + b.w) * w.w;
    }
    logits[e] = acc;
    atomicMax(&mkey[d], fkey(acc));
}

__global__ void k_gat_exp(const int* __restrict__ dst, float* __restrict__ logits,
                          const unsigned* __restrict__ mkey, float* __restrict__ ssum, int E) {
    int e = blockIdx.x * blockDim.x + threadIdx.x;
    if (e >= E) return;
    int d = dst[e];
    float ex = __expf(logits[e] - fkey_inv(mkey[d]));
    logits[e] = ex;                       // reuse buffer for exp values
    atomAddF32(&ssum[d], ex);
}

__global__ void k_gat_scatter(const int* __restrict__ src, const int* __restrict__ dst,
                              const float* __restrict__ ex, const float* __restrict__ ssum,
                              const float* __restrict__ fs, float* __restrict__ out, int E) {
    long t = (long)blockIdx.x * blockDim.x + threadIdx.x;
    if (t >= (long)E * DM) return;
    int e = (int)(t >> 5);
    int k = (int)(t & 31);
    int s = src[e], d = dst[e];
    float a = ex[e] / ssum[d];
    atomAddF32(out + (long)d * DM + k, a * fs[(long)s * DM + k]);
}

__global__ void k_bias_add(float* __restrict__ out, const float* __restrict__ bias, int N) {
    long t = (long)blockIdx.x * blockDim.x + threadIdx.x;
    if (t >= (long)N * DM) return;
    out[t] += bias[t & 31];
}

// ---------------- host orchestration ----------------
static inline int cdiv(long a, long b) { return (int)((a + b - 1) / b); }

extern "C" void kernel_launch(void* const* d_in, const int* in_sizes, int n_in,
                              void* d_out, int out_size, void* d_ws, size_t ws_size,
                              hipStream_t stream) {
    const float* x_in   = (const float*)d_in[0];
    const int*   src    = (const int*)d_in[1];
    const int*   dst    = (const int*)d_in[2];
    const float* lam    = (const float*)d_in[3];
    const float* chebW  = (const float*)d_in[4];   // (32, 96)
    const float* chebB  = (const float*)d_in[5];   // (32)
    const float* Wsrc   = (const float*)d_in[6];   // (32, 32)
    const float* bsrc   = (const float*)d_in[7];
    const float* Wdst   = (const float*)d_in[8];
    const float* bdst   = (const float*)d_in[9];
    const float* attn   = (const float*)d_in[10];  // (32)
    const float* gbias  = (const float*)d_in[11];

    const int N = in_sizes[0] / DM;
    const int E = in_sizes[1];
    float* out = (float*)d_out;

    // workspace carve-up (256B aligned)
    char* base = (char*)d_ws;
    size_t off = 0;
    auto alloc = [&](size_t bytes) -> char* {
        char* p = base + off;
        off = (off + bytes + 255) & ~(size_t)255;
        return p;
    };
    int*       deg   = (int*)      alloc((size_t)N * 4);
    float*     norm  = (float*)    alloc((size_t)N * 4);
    float*     hn    = (float*)    alloc((size_t)N * DM * 4);
    float*     p     = (float*)    alloc((size_t)N * DM * 4);
    float*     x1    = (float*)    alloc((size_t)N * DM * 4);
    _Float16*  xth   = (_Float16*) alloc((size_t)N * 96 * 2);
    float*     hA    = (float*)    alloc((size_t)N * DM * 4);   // cheb1 output (f32)
    float*     hB    = (float*)    alloc((size_t)N * DM * 4);   // cheb2 output (f32)
    _Float16*  h2h   = (_Float16*) alloc((size_t)N * DM * 2);   // half copy of GEMM input
    float*     fs    = (float*)    alloc((size_t)N * DM * 4);
    float*     fd    = (float*)    alloc((size_t)N * DM * 4);
    float*     lgt   = (float*)    alloc((size_t)E * 4);
    unsigned*  mkey  = (unsigned*) alloc((size_t)N * 4);
    float*     ssum  = (float*)    alloc((size_t)N * 4);
    _Float16*  wcH   = (_Float16*) alloc((size_t)32 * 96 * 2);
    _Float16*  wsH   = (_Float16*) alloc((size_t)32 * 32 * 2);
    _Float16*  wdH   = (_Float16*) alloc((size_t)32 * 32 * 2);

    const int B = 256;
    const long NW = (long)N * DM;   // node-feature elements
    const long EW = (long)E * DM;   // edge-feature elements

    // degree / norm
    hipMemsetAsync(deg, 0, (size_t)N * 4, stream);
    k_edge_deg<<<cdiv(E, B), B, 0, stream>>>(dst, deg, E);
    k_norm<<<cdiv(N, B), B, 0, stream>>>(deg, norm, N);

    // weights -> half (tiny)
    k_cast_half<<<cdiv(32 * 96, B), B, 0, stream>>>(chebW, wcH, 32 * 96);
    k_cast_half<<<cdiv(32 * 32, B), B, 0, stream>>>(Wsrc, wsH, 32 * 32);
    k_cast_half<<<cdiv(32 * 32, B), B, 0, stream>>>(Wdst, wdH, 32 * 32);

    const int gemmBlk = 128;                       // 4 waves
    const int gemmGrid = cdiv((long)(N >> 4), 4);  // one 16-row tile per wave

    // ---- Cheb layer (run twice, shared weights) ----
    auto cheb = [&](const float* xin, float* outF, _Float16* outH) {
        k_scale_rows<<<cdiv(NW, B), B, 0, stream>>>(xin, norm, hn, N);
        hipMemsetAsync(p, 0, (size_t)N * DM * 4, stream);
        k_edge_scatter<<<cdiv(EW, B), B, 0, stream>>>(src, dst, hn, p, E);
        k_cheb_x1<<<cdiv(NW, B), B, 0, stream>>>(xin, p, norm, lam, x1, hn, xth, N);
        hipMemsetAsync(p, 0, (size_t)N * DM * 4, stream);
        k_edge_scatter<<<cdiv(EW, B), B, 0, stream>>>(src, dst, hn, p, E);
        k_cheb_x2<<<cdiv(NW, B), B, 0, stream>>>(xin, x1, p, norm, lam, xth, N);
        k_gemm_wmma<<<gemmGrid, gemmBlk, 0, stream>>>(xth, wcH, chebB, outF, outH, N, 96, 1);
    };
    cheb(x_in, hA, h2h);   // h2h overwritten by second layer
    cheb(hA,  hB, h2h);

    // ---- GATv2 ----
    k_gemm_wmma<<<gemmGrid, gemmBlk, 0, stream>>>(h2h, wsH, bsrc, fs, nullptr, N, 32, 0);
    k_gemm_wmma<<<gemmGrid, gemmBlk, 0, stream>>>(h2h, wdH, bdst, fd, nullptr, N, 32, 0);

    hipMemsetAsync(mkey, 0, (size_t)N * 4, stream);   // key 0 == -inf under fkey order
    k_gat_logits<<<cdiv(E, B), B, 0, stream>>>(src, dst, fs, fd, attn, lgt, mkey, E);
    hipMemsetAsync(ssum, 0, (size_t)N * 4, stream);
    k_gat_exp<<<cdiv(E, B), B, 0, stream>>>(dst, lgt, mkey, ssum, E);
    hipMemsetAsync(out, 0, (size_t)N * DM * 4, stream);
    k_gat_scatter<<<cdiv(EW, B), B, 0, stream>>>(src, dst, lgt, ssum, fs, out, E);
    k_bias_add<<<cdiv(NW, B), B, 0, stream>>>(out, gbias, N);
}
